// Attention_54425825575424
// MI455X (gfx1250) — compile-verified, matrix operand order checked by reading, not run on hardware
//
#include <hip/hip_runtime.h>

#define N_TOK 8192
#define D 512
#define BM 32            /* query rows per block (2 strips of 16)           */
#define BN 128           /* keys per tile (8 waves x 16 keys)               */
#define SCALE 0.04419417382415922f  /* 1/sqrt(512) */

#define QS_STRIDE 528    /* 512 + 16 elems, 1056B: 16B-aligned, bank-skewed */
#define PS_STRIDE 144    /* 128 + 16 elems, 288B: 16B-aligned               */
#define SS_STRIDE 132    /* f32 scores, scalar access                       */

typedef __attribute__((ext_vector_type(16))) __bf16 v16bf;
typedef __attribute__((ext_vector_type(8)))  float  v8f;
typedef __attribute__((ext_vector_type(8)))  short  v8s;

struct bpair { v8s lo, hi; };

// 16x32-bf16 A/B fragment for v_wmma_f32_16x16x32_bf16 from two 16-byte
// chunks (K = kk..kk+7 and kk+16..kk+23 for this lane; lanes 0-15 kh=0,
// lanes 16-31 kh=1, per the ISA 16-bit A/B VGPR striping).
static __device__ __forceinline__ v16bf frag_from(const void* plo, const void* phi) {
  bpair f;
  f.lo = *(const v8s*)plo;
  f.hi = *(const v8s*)phi;
  return __builtin_bit_cast(v16bf, f);
}

// Hardware f32->bf16 convert (let clang pick v_cvt, ideally packed).
static __device__ __forceinline__ short f2bf(float x) {
  __bf16 h = (__bf16)x;
  return __builtin_bit_cast(short, h);
}

static __device__ __forceinline__ v8s cvt8(const float* __restrict__ p) {
  float4 a = *(const float4*)p;
  float4 b = *(const float4*)(p + 4);
  v8s r;
  r[0] = f2bf(a.x); r[1] = f2bf(a.y); r[2] = f2bf(a.z); r[3] = f2bf(a.w);
  r[4] = f2bf(b.x); r[5] = f2bf(b.y); r[6] = f2bf(b.z); r[7] = f2bf(b.w);
  return r;
}

static __device__ __forceinline__ v16bf frag_f32(const float* __restrict__ base) {
  bpair f;
  f.lo = cvt8(base);
  f.hi = cvt8(base + 16);
  return __builtin_bit_cast(v16bf, f);
}

// ---------------------------------------------------------------------------
// Kernel 1: Q = emb @ Wqk^T + bqk  (bf16, row-major [N][D])
//           V = emb @ Wv^T  + bv   (bf16, TRANSPOSED: Vt[D][N])
// One wave per 16x16 output tile; Q and V share the A (embedding) fragment.
// ---------------------------------------------------------------------------
__global__ __launch_bounds__(256) void proj_kernel(
    const float* __restrict__ emb,
    const float* __restrict__ Wqk, const float* __restrict__ bqk,
    const float* __restrict__ Wv,  const float* __restrict__ bv,
    short* __restrict__ Qg, short* __restrict__ Vt)
{
  const int tid  = threadIdx.x;
  const int lane = tid & 31;
  const int wave = tid >> 5;
  const int tile = blockIdx.x * 8 + wave;   // 512*32 = 16384 tiles
  const int tm = tile >> 5;                 // token tile   0..511
  const int tn = tile & 31;                 // feature tile 0..31
  const int am = lane & 15;
  const int kh = lane >> 4;

  const int row  = tm * 16 + am;            // A-lane: token row
  const int ocol = tn * 16 + am;            // B-lane: output feature

  v8f accQ = {};
  v8f accV = {};
  for (int s = 0; s < 16; ++s) {
    const int kk = s * 32 + kh * 8;
    v16bf a   = frag_f32(emb + row  * D + kk);
    v16bf bq  = frag_f32(Wqk + ocol * D + kk);
    v16bf bvv = frag_f32(Wv  + ocol * D + kk);
    accQ = __builtin_amdgcn_wmma_f32_16x16x32_bf16(false, a, false, bq,  (short)0, accQ, false, false);
    accV = __builtin_amdgcn_wmma_f32_16x16x32_bf16(false, a, false, bvv, (short)0, accV, false, false);
  }

  const float biasQ = bqk[ocol];
  const float biasV = bv[ocol];

  // D layout: VGPR r -> token row (tm*16 + kh*8 + r), column ocol.
  v8s vp;
  for (int r = 0; r < 8; ++r) {
    const int token = tm * 16 + kh * 8 + r;
    Qg[token * D + ocol] = f2bf(accQ[r] + biasQ);
    vp[r] = f2bf(accV[r] + biasV);
  }
  // Vt row = ocol, 8 contiguous tokens -> one 16B store.
  *(v8s*)(Vt + ocol * N_TOK + tm * 16 + kh * 8) = vp;
}

// ---------------------------------------------------------------------------
// Kernel 2: flash attention. Block = 32 query rows (2 strips), 8 waves.
// Per 128-key tile: wave w computes score subtiles (both strips) for keys
// [16w,16w+16) -- B fragment loaded ONCE, feeds TWO WMMAs; online softmax;
// then wave w computes P@V for output columns [64w, 64w+64) of both strips,
// again sharing each Vt B fragment across two WMMAs.
// ---------------------------------------------------------------------------
__global__ __launch_bounds__(256) void attn_kernel(
    const short* __restrict__ Qg, const short* __restrict__ Vt,
    float* __restrict__ out)
{
  __shared__ short Qs[BM * QS_STRIDE];      // 33.8 KB
  __shared__ float Ss[BM * SS_STRIDE];      // 16.9 KB
  __shared__ short Ps[BM * PS_STRIDE];      //  9.2 KB
  __shared__ float pmax[BM][8];
  __shared__ float psum[BM][8];
  __shared__ float m_run[BM], l_run[BM], m_new_s[BM], corr_s[BM];

  const int tid  = threadIdx.x;
  const int lane = tid & 31;
  const int wave = tid >> 5;
  const int am = lane & 15;
  const int kh = lane >> 4;
  const int qrow0 = blockIdx.x * BM;

  // Stage the 32x512 bf16 Q tile into LDS (2048 x 16B chunks, 8 per thread).
  for (int c = tid; c < BM * 64; c += 256) {
    const int r = c >> 6, cc = c & 63;
    *(v8s*)&Qs[r * QS_STRIDE + cc * 8] = *(const v8s*)(Qg + (qrow0 + r) * D + cc * 8);
  }
  if (tid < BM) { m_run[tid] = -3.0e38f; l_run[tid] = 0.0f; }
  __syncthreads();

  v8f acc0[4] = { {}, {}, {}, {} };   // strip 0 (rows 0..15)
  v8f acc1[4] = { {}, {}, {}, {} };   // strip 1 (rows 16..31)
  const int srow = tid >> 3;          // 0..31
  const int sseg = tid & 7;           // 16 cols per segment

  for (int kt = 0; kt < N_TOK / BN; ++kt) {
    const int kb = kt * BN;

    // ---- S subtiles: one key B-fragment feeds both strips' WMMAs
    v8f s0 = {}, s1 = {};
    const int keyrow = kb + wave * 16 + am;
    for (int s = 0; s < 16; ++s) {
      const int kk = s * 32 + kh * 8;
      v16bf b  = frag_from(Qg + keyrow * D + kk, Qg + keyrow * D + kk + 16);
      v16bf a0 = frag_from(&Qs[am * QS_STRIDE + kk],        &Qs[am * QS_STRIDE + kk + 16]);
      v16bf a1 = frag_from(&Qs[(16 + am) * QS_STRIDE + kk], &Qs[(16 + am) * QS_STRIDE + kk + 16]);
      s0 = __builtin_amdgcn_wmma_f32_16x16x32_bf16(false, a0, false, b, (short)0, s0, false, false);
      s1 = __builtin_amdgcn_wmma_f32_16x16x32_bf16(false, a1, false, b, (short)0, s1, false, false);
    }
    for (int r = 0; r < 8; ++r) {
      const int m = r + kh * 8;
      Ss[m * SS_STRIDE + wave * 16 + am]        = s0[r];
      Ss[(16 + m) * SS_STRIDE + wave * 16 + am] = s1[r];
    }
    __syncthreads();

    // ---- online softmax (256 threads: row = tid/8, 16 cols per thread)
    float mx = -3.0e38f;
    for (int j = 0; j < 16; ++j)
      mx = fmaxf(mx, Ss[srow * SS_STRIDE + sseg * 16 + j] * SCALE);
    pmax[srow][sseg] = mx;
    __syncthreads();
    if (sseg == 0) {
      float mt = pmax[srow][0];
      for (int j = 1; j < 8; ++j) mt = fmaxf(mt, pmax[srow][j]);
      const float mo = m_run[srow];
      const float mn = fmaxf(mo, mt);
      m_new_s[srow] = mn;
      corr_s[srow]  = __expf(mo - mn);
      m_run[srow]   = mn;
    }
    __syncthreads();
    {
      const float mn = m_new_s[srow];
      float ps = 0.0f;
      for (int j = 0; j < 16; ++j) {
        const int c = sseg * 16 + j;
        const float p = __expf(Ss[srow * SS_STRIDE + c] * SCALE - mn);
        ps += p;
        Ps[srow * PS_STRIDE + c] = f2bf(p);
      }
      psum[srow][sseg] = ps;
    }
    __syncthreads();
    if (sseg == 0) {
      float t = 0.0f;
      for (int j = 0; j < 8; ++j) t += psum[srow][j];
      l_run[srow] = l_run[srow] * corr_s[srow] + t;
    }

    // Warm WGP$ with next tile's key rows while PV WMMAs run.
    if (kb + BN < N_TOK)
      __builtin_prefetch(Qg + (kb + BN + wave * 16 + am) * D, 0, 1);

    // ---- O = corr*O + P @ V_tile ; wave w owns cols [64w, 64w+64), 2 strips
    float cr0[8], cr1[8];
    for (int r = 0; r < 8; ++r) {
      cr0[r] = corr_s[r + kh * 8];
      cr1[r] = corr_s[16 + r + kh * 8];
    }
    for (int j = 0; j < 4; ++j)
      for (int r = 0; r < 8; ++r) { acc0[j][r] *= cr0[r]; acc1[j][r] *= cr1[r]; }

    v16bf pa0[4], pa1[4];
    for (int s = 0; s < 4; ++s) {
      const int kk = s * 32 + kh * 8;
      pa0[s] = frag_from(&Ps[am * PS_STRIDE + kk],        &Ps[am * PS_STRIDE + kk + 16]);
      pa1[s] = frag_from(&Ps[(16 + am) * PS_STRIDE + kk], &Ps[(16 + am) * PS_STRIDE + kk + 16]);
    }
    for (int j = 0; j < 4; ++j) {
      const int ncol = wave * 64 + j * 16 + am;
      const short* vb = Vt + ncol * N_TOK + kb;   // Vt row-major: contiguous K
      for (int s = 0; s < 4; ++s) {
        const int kk = s * 32 + kh * 8;
        v16bf b = frag_from(vb + kk, vb + kk + 16);
        acc0[j] = __builtin_amdgcn_wmma_f32_16x16x32_bf16(false, pa0[s], false, b, (short)0, acc0[j], false, false);
        acc1[j] = __builtin_amdgcn_wmma_f32_16x16x32_bf16(false, pa1[s], false, b, (short)0, acc1[j], false, false);
      }
    }
    __syncthreads();
  }

  // ---- epilogue: O /= l, store f32
  float li0[8], li1[8];
  for (int r = 0; r < 8; ++r) {
    li0[r] = 1.0f / l_run[r + kh * 8];
    li1[r] = 1.0f / l_run[16 + r + kh * 8];
  }
  for (int j = 0; j < 4; ++j) {
    const int ncol = wave * 64 + j * 16 + am;
    for (int r = 0; r < 8; ++r) {
      const int m = r + kh * 8;
      out[(qrow0 + m) * D + ncol]      = acc0[j][r] * li0[r];
      out[(qrow0 + 16 + m) * D + ncol] = acc1[j][r] * li1[r];
    }
  }
}

extern "C" void kernel_launch(void* const* d_in, const int* in_sizes, int n_in,
                              void* d_out, int out_size, void* d_ws, size_t ws_size,
                              hipStream_t stream) {
  const float* emb = (const float*)d_in[0];
  const float* Wqk = (const float*)d_in[1];
  const float* bqk = (const float*)d_in[2];
  const float* Wv  = (const float*)d_in[3];
  const float* bv  = (const float*)d_in[4];

  short* Qg = (short*)d_ws;                       // bf16 [8192][512]  (8 MB)
  short* Vt = Qg + (size_t)N_TOK * D;             // bf16 [512][8192]  (8 MB)
  float* out = (float*)d_out;                     // f32  [8192][512]

  proj_kernel<<<2048, 256, 0, stream>>>(emb, Wqk, bqk, Wv, bv, Qg, Vt);
  attn_kernel<<<N_TOK / BM, 256, 0, stream>>>(Qg, Vt, out);
}
